// OfflineSlidingWindowAttn_38603166056861
// MI455X (gfx1250) — matrix operand order repaired and checked by reading, not compiled, not run on hardware
//
#include <hip/hip_runtime.h>

typedef __attribute__((ext_vector_type(16))) _Float16 v16h;
typedef __attribute__((ext_vector_type(8)))  float    v8f;

namespace {
constexpr int   kB    = 2;
constexpr int   kS    = 2048;
constexpr int   kHQ   = 16;
constexpr int   kHKV  = 4;
constexpr int   kD    = 128;
constexpr int   kWin  = 512;
constexpr float kEps  = 1e-6f;
constexpr float kScale = 0.08838834764831845f;   // 1/sqrt(128)

constexpr int kQSub  = 16;            // rows per WMMA sub-tile
constexpr int kQRows = 2 * kQSub;     // 32 query rows per wave (2 sub-tiles)
constexpr int kKTile = 32;            // keys per iteration
constexpr int kWaves = 2;             // waves per block
constexpr int kBlk   = 32 * kWaves;

// per-wave LDS (halves): Q(32x128) + K(32x128) + V(32x128) + P(32x32)
constexpr int kQH = kQRows * kD;
constexpr int kKH = kKTile * kD;
constexpr int kVH = kKTile * kD;
constexpr int kPH = kQRows * kKTile;
constexpr int kWaveHalves = kQH + kKH + kVH + kPH;     // 13312 halves
constexpr size_t kSmemBytes =
    2 * kD * sizeof(float) + (size_t)kWaves * kWaveHalves * sizeof(_Float16); // 54272 B
}

// Compiler-only ordering barrier for aliasing LDS access sequences.
// The LDS pipe is in-order per wave (store->load forwarding is guaranteed by
// hardware), and the compiler inserts minimal partial s_wait_dscnt before
// consuming loaded registers — so no full-drain hardware wait is needed.
#define LDS_ORDER() asm volatile("" ::: "memory")

// Online-softmax update for one 16x32 logit tile held as two 16x16 C-fragments.
// Row = vgpr r + 8*hi, col = lane&15 (+16 for sf1). kMasked=false is the
// interior fast path (tile fully inside the sliding window; wave-uniform).
template <bool kMasked>
__device__ __forceinline__ void softmax_update(int qb, int j0, int m, int hi,
                                               const v8f& sf0, const v8f& sf1,
                                               float* __restrict__ mrow,
                                               float* __restrict__ lrow,
                                               v8f* __restrict__ acc,
                                               _Float16* __restrict__ prow) {
#pragma unroll
  for (int r = 0; r < 8; ++r) {
    const int ig = qb + hi * 8 + r;
    bool ok0 = true, ok1 = true;
    if (kMasked) {
      const int jg0 = j0 + m, jg1 = jg0 + 16;
      ok0 = (jg0 <= ig) && (ig - jg0 <= kWin);
      ok1 = (jg1 <= ig) && (ig - jg1 <= kWin);
    }
    const float s0v = sf0[r];
    const float s1v = sf1[r];
    float cand = fmaxf(ok0 ? s0v : -1e30f, ok1 ? s1v : -1e30f);
    cand = fmaxf(cand, __shfl_xor(cand, 1));
    cand = fmaxf(cand, __shfl_xor(cand, 2));
    cand = fmaxf(cand, __shfl_xor(cand, 4));
    cand = fmaxf(cand, __shfl_xor(cand, 8));
    const float mn = fmaxf(mrow[r], cand);
    const float al = __expf(mrow[r] - mn);
    const float p0 = ok0 ? __expf(s0v - mn) : 0.f;
    const float p1 = ok1 ? __expf(s1v - mn) : 0.f;
    float ps = p0 + p1;
    ps += __shfl_xor(ps, 1);
    ps += __shfl_xor(ps, 2);
    ps += __shfl_xor(ps, 4);
    ps += __shfl_xor(ps, 8);
    lrow[r] = lrow[r] * al + ps;
    mrow[r] = mn;
#pragma unroll
    for (int t = 0; t < 8; ++t) acc[t][r] *= al;
    prow[(hi * 8 + r) * kKTile + m]      = (_Float16)p0;
    prow[(hi * 8 + r) * kKTile + 16 + m] = (_Float16)p1;
  }
}

__global__ __launch_bounds__(kBlk)
void swa_fa_kernel(const float* __restrict__ qg_ptr,
                   const float* __restrict__ kg_ptr,
                   const float* __restrict__ vg_ptr,
                   const float* __restrict__ qgamma,
                   const float* __restrict__ kgamma,
                   float* __restrict__ out) {
  const int tid  = threadIdx.x;
  const int lane = tid & 31;
  const int wid  = tid >> 5;
  const int m    = lane & 15;
  const int hi   = lane >> 4;

  const int blk = blockIdx.x;
  const int qg  = blk & 31;            // 32 groups of (kWaves*32)=64 query rows
  const int h   = (blk >> 5) & 15;
  const int b   = blk >> 9;
  const int q0  = (qg * kWaves + wid) * kQRows;
  const int kvh = h >> 2;              // GQA: 4 q-heads per kv-head

  extern __shared__ char smem_raw[];
  float* gq_s = reinterpret_cast<float*>(smem_raw);
  float* gk_s = gq_s + kD;
  _Float16* wbase = reinterpret_cast<_Float16*>(smem_raw + 2 * kD * sizeof(float))
                    + (size_t)wid * kWaveHalves;
  _Float16* q_lds = wbase;
  _Float16* k_lds = q_lds + kQH;
  _Float16* v_lds = k_lds + kKH;
  _Float16* p_lds = v_lds + kVH;

  // --- per-head gamma slices to LDS (same h for both waves) ---
  for (int i = tid; i < kD; i += kBlk) {
    gq_s[i] = qgamma[h * kD + i];
    gk_s[i] = kgamma[h * kD + i];
  }
  __syncthreads();

  // --- Q tiles: GroupRMSNorm + gamma + softmax scale -> f16 LDS (2 sub-passes) ---
#pragma unroll
  for (int g = 0; g < 2; ++g) {
    const int row = g * kQSub + m;
    const float4* qp4 = reinterpret_cast<const float4*>(
        qg_ptr + ((((size_t)b * kS) + (q0 + row)) * kHQ + h) * kD + hi * 64);
    float ss = 0.f;
#pragma unroll
    for (int i = 0; i < 16; ++i) {
      float4 x = qp4[i];
      ss += x.x * x.x + x.y * x.y + x.z * x.z + x.w * x.w;
    }
    ss += __shfl_xor(ss, 16);  // join the two d-halves of the row
    const float inv = rsqrtf(ss * (1.f / 128.f) + kEps) * kScale;
#pragma unroll
    for (int i = 0; i < 16; ++i) {
      float4 x = qp4[i];
      const int d = hi * 64 + i * 4;
      q_lds[row * kD + d + 0] = (_Float16)(x.x * inv * gq_s[d + 0]);
      q_lds[row * kD + d + 1] = (_Float16)(x.y * inv * gq_s[d + 1]);
      q_lds[row * kD + d + 2] = (_Float16)(x.z * inv * gq_s[d + 2]);
      q_lds[row * kD + d + 3] = (_Float16)(x.w * inv * gq_s[d + 3]);
    }
  }
  LDS_ORDER();

  // --- 8 persistent A fragments (2 sub-tiles x 4 d-chunks; ISA 16-bit A layout) ---
  v16h qA[2][4];
#pragma unroll
  for (int g = 0; g < 2; ++g) {
#pragma unroll
    for (int c = 0; c < 4; ++c) {
#pragma unroll
      for (int e = 0; e < 16; ++e) {
        const int vv = e >> 1, hh = e & 1;
        const int kk = ((vv & 4) ? 16 : 0) + (hi ? 8 : 0) + ((vv & 3) << 1) + hh;
        qA[g][c][e] = q_lds[(g * kQSub + m) * kD + c * 32 + kk];
      }
    }
  }

  // --- flash-attention state: O is 32x128 f32 across 16 C-fragments ---
  const v8f z = {};
  v8f acc[2][8];
  float mrow[2][8], lrow[2][8];
#pragma unroll
  for (int g = 0; g < 2; ++g)
#pragma unroll
    for (int t = 0; t < 8; ++t) { acc[g][t] = z; mrow[g][t] = -1e30f; lrow[g][t] = 0.f; }

  int jstart = q0 - kWin; if (jstart < 0) jstart = 0;
  jstart &= ~(kKTile - 1);
  const int jlast = q0 + kQRows - 1;

  for (int j0 = jstart; j0 <= jlast; j0 += kKTile) {
    // --- prefetch next K/V tile rows (global_prefetch_b8; overlaps HBM latency) ---
    if (j0 + kKTile <= q0) {  // next iteration exists and rows stay in-range
      const size_t nrow = ((size_t)b * kS + (j0 + kKTile + lane)) * kHKV + kvh;
      __builtin_prefetch(kg_ptr + nrow * kD, 0, 3);
      __builtin_prefetch(vg_ptr + nrow * kD, 0, 3);
    }

    // --- stage K (RMS-norm * gamma) and V (f32->f16): one row per lane ---
    {
      const int krow = j0 + lane;  // always in [0, S)
      const float4* kp4 = reinterpret_cast<const float4*>(
          kg_ptr + ((((size_t)b * kS) + krow) * kHKV + kvh) * kD);
      float ss = 0.f;
#pragma unroll
      for (int i = 0; i < 32; ++i) {
        float4 x = kp4[i];
        ss += x.x * x.x + x.y * x.y + x.z * x.z + x.w * x.w;
      }
      const float kinv = rsqrtf(ss * (1.f / 128.f) + kEps);
#pragma unroll
      for (int i = 0; i < 32; ++i) {
        float4 x = kp4[i];
        const int d = i * 4;
        k_lds[lane * kD + d + 0] = (_Float16)(x.x * kinv * gk_s[d + 0]);
        k_lds[lane * kD + d + 1] = (_Float16)(x.y * kinv * gk_s[d + 1]);
        k_lds[lane * kD + d + 2] = (_Float16)(x.z * kinv * gk_s[d + 2]);
        k_lds[lane * kD + d + 3] = (_Float16)(x.w * kinv * gk_s[d + 3]);
      }
      const float4* vp4 = reinterpret_cast<const float4*>(
          vg_ptr + ((((size_t)b * kS) + krow) * kHKV + kvh) * kD);
#pragma unroll
      for (int i = 0; i < 32; ++i) {
        float4 x = vp4[i];
        const int d = i * 4;
        v_lds[lane * kD + d + 0] = (_Float16)x.x;
        v_lds[lane * kD + d + 1] = (_Float16)x.y;
        v_lds[lane * kD + d + 2] = (_Float16)x.z;
        v_lds[lane * kD + d + 3] = (_Float16)x.w;
      }
    }
    LDS_ORDER();

    // --- S = Qn*Kn^T: 4 C-fragments (2 q-sub-tiles x 2 key halves), shared B loads ---
    v8f s00 = z, s01 = z, s10 = z, s11 = z;
#pragma unroll
    for (int c = 0; c < 4; ++c) {
      v16h b0, b1;
#pragma unroll
      for (int e = 0; e < 16; ++e) {
        const int dd = c * 32 + ((e & 8) << 1) + hi * 8 + (e & 7);  // B layout: k = 16*(e>=8) + 8*hi + (e&7), col = m
        b0[e] = k_lds[m * kD + dd];
        b1[e] = k_lds[(16 + m) * kD + dd];
      }
      s00 = __builtin_amdgcn_wmma_f32_16x16x32_f16(false, qA[0][c], false, b0,
                                                   (short)0, s00, false, false);
      s10 = __builtin_amdgcn_wmma_f32_16x16x32_f16(false, qA[1][c], false, b0,
                                                   (short)0, s10, false, false);
      s01 = __builtin_amdgcn_wmma_f32_16x16x32_f16(false, qA[0][c], false, b1,
                                                   (short)0, s01, false, false);
      s11 = __builtin_amdgcn_wmma_f32_16x16x32_f16(false, qA[1][c], false, b1,
                                                   (short)0, s11, false, false);
    }

    // --- online softmax per sub-tile; interior tiles take the maskless path ---
#pragma unroll
    for (int g = 0; g < 2; ++g) {
      const int qb = q0 + g * kQSub;
      const v8f& f0 = g ? s10 : s00;
      const v8f& f1 = g ? s11 : s01;
      _Float16* prow = p_lds + g * kQSub * kKTile;
      const bool full = (j0 + kKTile - 1 <= qb) && (qb + kQSub - 1 - j0 <= kWin);
      if (full)
        softmax_update<false>(qb, j0, m, hi, f0, f1, mrow[g], lrow[g], acc[g], prow);
      else
        softmax_update<true>(qb, j0, m, hi, f0, f1, mrow[g], lrow[g], acc[g], prow);
    }
    LDS_ORDER();

    // --- O += P(32x32) * V(32x128): shared V B-fragments feed both sub-tiles ---
    v16h pA[2];
#pragma unroll
    for (int g = 0; g < 2; ++g) {
#pragma unroll
      for (int e = 0; e < 16; ++e) {
        const int vv = e >> 1, hh = e & 1;
        const int kk = ((vv & 4) ? 16 : 0) + (hi ? 8 : 0) + ((vv & 3) << 1) + hh;
        pA[g][e] = p_lds[(g * kQSub + m) * kKTile + kk];
      }
    }
#pragma unroll
    for (int t = 0; t < 8; ++t) {
      v16h vb;
#pragma unroll
      for (int e = 0; e < 16; ++e)
        vb[e] = v_lds[(((e & 8) << 1) + hi * 8 + (e & 7)) * kD + t * 16 + m];  // B: k = key, n = d-lane
      acc[0][t] = __builtin_amdgcn_wmma_f32_16x16x32_f16(false, pA[0], false, vb,
                                                         (short)0, acc[0][t], false, false);
      acc[1][t] = __builtin_amdgcn_wmma_f32_16x16x32_f16(false, pA[1], false, vb,
                                                         (short)0, acc[1][t], false, false);
    }
    LDS_ORDER();  // keep next tile's LDS overwrite ordered behind these reads
  }

  // --- normalize (one rcp per row); non-temporal stores (output never re-read) ---
#pragma unroll
  for (int g = 0; g < 2; ++g) {
    float inv[8];
#pragma unroll
    for (int r = 0; r < 8; ++r) inv[r] = 1.0f / lrow[g][r];
#pragma unroll
    for (int t = 0; t < 8; ++t) {
#pragma unroll
      for (int r = 0; r < 8; ++r) {
        const int ig = q0 + g * kQSub + hi * 8 + r;
        float* dst = out + ((((size_t)b * kS) + ig) * kHQ + h) * kD + t * 16 + m;
        __builtin_nontemporal_store(acc[g][t][r] * inv[r], dst);
      }
    }
  }
}

extern "C" void kernel_launch(void* const* d_in, const int* in_sizes, int n_in,
                              void* d_out, int out_size, void* d_ws, size_t ws_size,
                              hipStream_t stream) {
  (void)in_sizes; (void)n_in; (void)out_size; (void)d_ws; (void)ws_size;
  const float* q      = (const float*)d_in[0];
  const float* k      = (const float*)d_in[1];
  const float* v      = (const float*)d_in[2];
  const float* qgamma = (const float*)d_in[3];
  const float* kgamma = (const float*)d_in[4];
  float* out = (float*)d_out;

  const int grid = kB * kHQ * (kS / (kQRows * kWaves));  // 1024 blocks
  hipLaunchKernelGGL(swa_fa_kernel, dim3(grid), dim3(kBlk), kSmemBytes, stream,
                     q, k, v, qgamma, kgamma, out);
}
